// ODConv1D_10703058502156
// MI455X (gfx1250) — matrix operand order repaired and checked by reading, not compiled, not run on hardware
//
#include <hip/hip_runtime.h>
#include <stdint.h>

#define Bq 8
#define CIN 512
#define COUT 512
#define KW 5
#define Lx 4096
#define REDc 128
#define KDIM (CIN * KW)        // 2560
#define KSTEPS (KDIM / 32)     // 80 (even)

typedef __attribute__((ext_vector_type(16))) __bf16    v16bf;
typedef __attribute__((ext_vector_type(8)))  float     v8f;
typedef __attribute__((ext_vector_type(4)))  unsigned  u32x4;

__device__ __forceinline__ float sigmoidf(float x) { return 1.0f / (1.0f + __expf(-x)); }

// ---------------- kernel 1: mean over L ----------------
__global__ void k_mean(const float* __restrict__ x, float* __restrict__ avg) {
  const int bc = blockIdx.x;                       // b*CIN + c
  const float* p = x + (size_t)bc * Lx;
  float s = 0.f;
  for (int i = threadIdx.x; i < Lx; i += 256) s += p[i];
  __shared__ float red[256];
  red[threadIdx.x] = s;
  __syncthreads();
  for (int off = 128; off > 0; off >>= 1) {
    if ((int)threadIdx.x < off) red[threadIdx.x] += red[threadIdx.x + off];
    __syncthreads();
  }
  if (threadIdx.x == 0) avg[bc] = red[0] * (1.0f / Lx);
}

// ---------------- kernel 2: four attention MLPs (one block per batch) ----------------
__global__ void k_att(const float* __restrict__ avg,
                      const float* kw1, const float* kb1, const float* kw2, const float* kb2,
                      const float* sw1, const float* sb1, const float* sw2, const float* sb2,
                      const float* iw1, const float* ib1, const float* iw2, const float* ib2,
                      const float* ow1, const float* ob1, const float* ow2, const float* ob2,
                      float* __restrict__ ka_out, float* __restrict__ s_out) {
  const int b = blockIdx.x, tid = threadIdx.x;
  __shared__ float a_sm[CIN];
  __shared__ float h[4][REDc];
  __shared__ float sa_sh;
  for (int i = tid; i < CIN; i += 256) a_sm[i] = avg[b * CIN + i];
  __syncthreads();
  const float* w1s[4] = {kw1, sw1, iw1, ow1};
  const float* b1s[4] = {kb1, sb1, ib1, ob1};
  for (int t = tid; t < 4 * REDc; t += 256) {
    int br = t >> 7, j = t & 127;
    const float* w = w1s[br] + j * CIN;
    float d = b1s[br][j];
    for (int i = 0; i < CIN; ++i) d += w[i] * a_sm[i];
    h[br][j] = d > 0.f ? d : 0.f;                   // relu
  }
  __syncthreads();
  if (tid < 6) {
    if (tid == 5) {                                  // sa (scalar)
      float d = sb2[0];
      for (int j = 0; j < REDc; ++j) d += h[1][j] * sw2[j];
      sa_sh = sigmoidf(d);
    } else {                                         // ka[0..4]
      float d = kb2[tid];
      for (int j = 0; j < REDc; ++j) d += h[0][j] * kw2[tid * REDc + j];
      ka_out[b * KW + tid] = sigmoidf(d);
    }
  }
  __syncthreads();
  for (int o = tid; o < COUT; o += 256) {            // s[b,o] = sa*ia*oa
    float di = ib2[o], dz = ob2[o];
    for (int j = 0; j < REDc; ++j) {
      di += h[2][j] * iw2[o * REDc + j];
      dz += h[3][j] * ow2[o * REDc + j];
    }
    s_out[b * COUT + o] = sa_sh * sigmoidf(di) * sigmoidf(dz);
  }
}

// ---------------- kernel 3: prebuild per-batch bf16 A fragments (WMMA A 16x32 layout) ----
// Afrag layout (uint = 2 bf16): idx = (((b*KSTEPS+ks)*32 + mb)*32 + lane)*8 + j
__global__ void k_prepA(const float* __restrict__ weight, const float* __restrict__ ka,
                        uint32_t* __restrict__ Afrag) {
  unsigned idx  = blockIdx.x * 256u + threadIdx.x;
  unsigned j    =  idx        & 7u;
  unsigned lane = (idx >> 3)  & 31u;
  unsigned mb   = (idx >> 8)  & 31u;
  unsigned t    =  idx >> 13;
  unsigned ks   = t % KSTEPS;
  unsigned b    = t / KSTEPS;
  unsigned o    = mb * 16u + (lane & 15u);
  union { __bf16 h[2]; unsigned u; } pk;
#pragma unroll
  for (int e = 0; e < 2; ++e) {
    unsigned hi  = 2u * j + (unsigned)e;
    // A-layout: contiguous K, halves 0..15 -> K 0..15; lanes>=16 shift K by 16
    unsigned kkl = (lane >> 4) * 16u + hi;
    unsigned kkg = ks * 32u + kkl;                   // kk = k*512 + i
    unsigned k = kkg >> 9, i = kkg & 511u;
    pk.h[e] = (__bf16)(weight[(o * CIN + i) * KW + k] * ka[b * KW + k]);
  }
  Afrag[idx] = pk.u;
}

// ---------------- kernel 4: batched implicit-GEMM conv via WMMA bf16 ----------------
__global__ __launch_bounds__(256)
void k_conv(const float* __restrict__ x, const u32x4* __restrict__ Afrag,
            const float* __restrict__ s, const float* __restrict__ bias,
            float* __restrict__ out) {
  const int tid  = threadIdx.x;
  const int lane = tid & 31, wave = tid >> 5;
  const int wm = wave >> 2, wn = wave & 3;           // 2 x 4 wave grid over 128x128 tile
  const int b  = blockIdx.z;
  const int n0 = blockIdx.x * 128;
  const int m0blk = blockIdx.y * 8;                  // global 16-row block base

  __shared__ __attribute__((aligned(16))) unsigned short Bt[2][32 * 128]; // 2 x 8KB, frag layout
  __shared__ float s_sm[128], bias_sm[128];

  if (tid < 128) {
    int o = blockIdx.y * 128 + tid;
    s_sm[tid]    = s[b * COUT + o];
    bias_sm[tid] = bias[o];
  }

  union Frag { u32x4 u[2]; v16bf v; };

  v8f acc[4][2];
#pragma unroll
  for (int f = 0; f < 4; ++f)
#pragma unroll
    for (int q = 0; q < 2; ++q)
#pragma unroll
      for (int r = 0; r < 8; ++r) acc[f][q][r] = 0.f;

  // ---- loop-invariant staging indexing (two chunks per thread: g0 and g0+2) ----
  const int nn    = tid & 127;                       // column within tile (same for both chunks)
  const int g0    = tid >> 7;                        // K-group-of-8: chunk0 = g0, chunk1 = g0+2
  const int lbase = n0 + nn - 2;                     // PAD = 2
  const int bCIN  = b * CIN;
  // LDS destinations (fragment-ready): chunk1 sits 16 lanes (256 halves) above chunk0
  const int dbase = (nn >> 4) * 512 + (nn & 15) * 16 + (g0 & 1) * 8;

  // stage one 32(K) x 128(N) slice of im2col(x): unconditional clamped loads * mask
  auto stageB = [&](int p, int ks) {
    const int k  = ks >> 4;                          // kernel tap (constant over slice)
    const int i0 = (ks & 15) * 32;                   // input-channel base
    const int l  = lbase + k;
    const float msk = (l >= 0 && l < Lx) ? 1.f : 0.f;
    const int lc = l < 0 ? 0 : (l >= Lx ? Lx - 1 : l);
    const float* xp = x + ((size_t)(bCIN + i0 + g0 * 8) * Lx) + lc;
#pragma unroll
    for (int cc = 0; cc < 2; ++cc) {                 // rows (g0+2*cc)*8 .. +7
      union { __bf16 h[8]; u32x4 u; } pk;
#pragma unroll
      for (int jj = 0; jj < 8; ++jj)
        pk.h[jj] = (__bf16)(xp[(size_t)(cc * 16 + jj) * Lx] * msk);
      *(u32x4*)&Bt[p][dbase + cc * 256] = pk.u;
    }
  };

  // A fragments straight from L2 (prebuilt, fragment-ready)
  auto loadA = [&](Frag (&a)[4], int ks) {
    const size_t kbase = ((size_t)(b * KSTEPS + ks)) * 32;
#pragma unroll
    for (int f = 0; f < 4; ++f) {
      size_t u4i = ((kbase + (size_t)(m0blk + wm * 4 + f)) * 32 + lane) * 2;
      a[f].u[0] = Afrag[u4i];
      a[f].u[1] = Afrag[u4i + 1];
    }
  };

  // compute one K-step: B fragments from LDS buffer p, A fragments from registers
  auto compute = [&](const Frag (&a)[4], int p) {
    Frag bf[2];
#pragma unroll
    for (int q = 0; q < 2; ++q) {
      const u32x4* bp = (const u32x4*)&Bt[p][(wn * 2 + q) * 512 + lane * 16];
      bf[q].u[0] = bp[0];
      bf[q].u[1] = bp[1];
    }
#pragma unroll
    for (int f = 0; f < 4; ++f)
#pragma unroll
      for (int q = 0; q < 2; ++q)
        acc[f][q] = __builtin_amdgcn_wmma_f32_16x16x32_bf16(
            false, a[f].v, false, bf[q].v, (short)0, acc[f][q], false, false);
  };

  // ---- software pipeline: A ping-pongs in registers (no copies), B in LDS buffers ----
  Frag aP[4], aQ[4];
  loadA(aP, 0);
  stageB(0, 0);
  __syncthreads();

  for (int ks = 0; ks < KSTEPS; ks += 2) {
    // even half-step: compute with aP / Bt[0]; prefetch ks+1 into aQ / Bt[1]
    loadA(aQ, ks + 1);                               // ks+1 <= 79 always (KSTEPS even)
    stageB(1, ks + 1);
    compute(aP, 0);
    __syncthreads();

    // odd half-step: compute with aQ / Bt[1]; prefetch ks+2 into aP / Bt[0]
    if (ks + 2 < KSTEPS) {
      loadA(aP, ks + 2);
      stageB(0, ks + 2);
    }
    compute(aQ, 1);
    __syncthreads();
  }

  // epilogue: out = s[b,o] * acc + bias[o]  (C/D layout: row = r + 8*(lane/16), col = lane%16)
#pragma unroll
  for (int f = 0; f < 4; ++f) {
    int olo_base = (wm * 4 + f) * 16 + ((lane >> 4) << 3);
#pragma unroll
    for (int q = 0; q < 2; ++q) {
      int n = n0 + (wn * 2 + q) * 16 + (lane & 15);
#pragma unroll
      for (int r = 0; r < 8; ++r) {
        int olo = olo_base + r;
        int o   = blockIdx.y * 128 + olo;
        out[((size_t)(b * COUT + o)) * Lx + n] = acc[f][q][r] * s_sm[olo] + bias_sm[olo];
      }
    }
  }
}

// ---------------- host launcher ----------------
extern "C" void kernel_launch(void* const* d_in, const int* in_sizes, int n_in,
                              void* d_out, int out_size, void* d_ws, size_t ws_size,
                              hipStream_t stream) {
  const float* x      = (const float*)d_in[0];
  const float* weight = (const float*)d_in[1];
  const float* bias   = (const float*)d_in[2];
  const float* kw1 = (const float*)d_in[3],  *kb1 = (const float*)d_in[4];
  const float* kw2 = (const float*)d_in[5],  *kb2 = (const float*)d_in[6];
  const float* sw1 = (const float*)d_in[7],  *sb1 = (const float*)d_in[8];
  const float* sw2 = (const float*)d_in[9],  *sb2 = (const float*)d_in[10];
  const float* iw1 = (const float*)d_in[11], *ib1 = (const float*)d_in[12];
  const float* iw2 = (const float*)d_in[13], *ib2 = (const float*)d_in[14];
  const float* ow1 = (const float*)d_in[15], *ob1 = (const float*)d_in[16];
  const float* ow2 = (const float*)d_in[17], *ob2 = (const float*)d_in[18];

  float*    avg   = (float*)d_ws;                               // 4096 f32
  float*    s_ws  = avg + Bq * CIN;                             // 4096 f32
  float*    ka_ws = s_ws + Bq * COUT;                           // 40 f32
  uint32_t* Afrag = (uint32_t*)((char*)d_ws + 64 * 1024);       // 8*80*32*32*8 uints = 21MB

  k_mean<<<dim3(Bq * CIN), 256, 0, stream>>>(x, avg);
  k_att<<<dim3(Bq), 256, 0, stream>>>(avg,
      kw1, kb1, kw2, kb2, sw1, sb1, sw2, sb2,
      iw1, ib1, iw2, ib2, ow1, ob1, ow2, ob2, ka_ws, s_ws);
  const unsigned nAuints = (unsigned)Bq * KSTEPS * 32 * 32 * 8; // 5,242,880
  k_prepA<<<dim3(nAuints / 256), 256, 0, stream>>>(weight, ka_ws, Afrag);
  k_conv<<<dim3(Lx / 128, COUT / 128, Bq), 256, 0, stream>>>(
      x, (const u32x4*)Afrag, s_ws, bias, (float*)d_out);
}
